// Generator_61753039782670
// MI455X (gfx1250) — matrix operand order repaired and checked by reading, not compiled, not run on hardware
//
#include <hip/hip_runtime.h>

// ---------------- problem constants ----------------
#define HH     200          // LSTM hidden size
#define H4     800          // 4*H (gate concat width)
#define NFC    400          // dense head output width
#define KP     224          // K padded to multiple of 32 (7 * 32)
#define KT     7            // K tiles of 32
#define TSEQ   100          // sequence length
#define MT     32           // batch rows per block (two WMMA M tiles)
#define NT4H   50           // N tiles of 16 covering 800
#define NTFC   25           // N tiles of 16 covering 400
#define THREADS 512         // 16 waves (wave32)
#define NTSTRIDE 8          // nt streams = WAVES/2 (two m-tiles per nt stream)

typedef __bf16 bf16_t;
typedef __bf16 v16bf __attribute__((ext_vector_type(16)));
typedef __bf16 v8bf  __attribute__((ext_vector_type(8)));
typedef float  v8f   __attribute__((ext_vector_type(8)));

// ---------------- weight pack: fp32 [Ksrc,N] -> bf16 [KP,N], zero pad K ----------------
__global__ void lstm_pack_w(const float* __restrict__ src, bf16_t* __restrict__ dst,
                            int Ksrc, int N)
{
    int i = blockIdx.x * blockDim.x + threadIdx.x;
    int total = KP * N;
    if (i >= total) return;
    int k = i / N;
    int n = i - k * N;
    float v = (k < Ksrc) ? src[k * N + n] : 0.0f;
    dst[i] = (bf16_t)v;
}

// ---------------- activation helpers (exp-based -> v_exp_f32 trans ops) ----------------
__device__ __forceinline__ float sigm_f(float x)
{
    return 1.0f / (1.0f + __expf(-x));
}
__device__ __forceinline__ float tanh_f(float x)
{
    x = fminf(15.0f, fmaxf(-15.0f, x));
    float e = __expf(2.0f * x);
    return (e - 1.0f) / (e + 1.0f);
}

// ---------------- WMMA fragment helpers ----------------
// A (16x32 bf16, M x K) from LDS h-buffer stored row-major [MT][KP].
// ISA layout: lanes 0-15 hold M=0-15 with K {0..7, 16..23}; lanes 16-31 K {8..15, 24..31}.
// shL points at the first row of the 16-row m-tile.
__device__ __forceinline__ v16bf load_a_frag(const bf16_t* shL, int kt, int lane)
{
    int m  = lane & 15;
    int kb = (lane >> 4) << 3;                    // 0 or 8
    const bf16_t* p = shL + m * KP + kt * 32;
    v8bf lo = *(const v8bf*)(p + kb);
    v8bf hi = *(const v8bf*)(p + kb + 16);
    return __builtin_shufflevector(lo, hi, 0,1,2,3,4,5,6,7,8,9,10,11,12,13,14,15);
}

// B (32x16 bf16, K x N) from global row-major packed weight [KP][N].
// lane l holds column N = nt*16 + l%16, K {0..7, 16..23} (lanes 0-15) or {8..15, 24..31} (lanes 16-31).
__device__ __forceinline__ v16bf load_b_frag(const bf16_t* __restrict__ w, int N,
                                             int kt, int nt, int lane)
{
    int n  = nt * 16 + (lane & 15);
    int kb = kt * 32 + ((lane >> 4) << 3);
    v16bf b;
#pragma unroll
    for (int s = 0; s < 16; ++s)
        b[s] = w[(kb + (s < 8 ? s : s + 8)) * N + n];
    return b;
}

// C/D (16x16 f32): VGPR r, lane l -> M = r + (l/16)*8, N = l%16.
// szRow points at the first row of the m-tile inside sz.
__device__ __forceinline__ void store_c_lds(float* szRow, int N, int nt, int lane, v8f acc)
{
    int n  = nt * 16 + (lane & 15);
    int mb = (lane >> 4) << 3;
#pragma unroll
    for (int r = 0; r < 8; ++r)
        szRow[(mb + r) * N + n] = acc[r];
}

// ---------------- gate pass: z[MT][800] + bias (+ x*W0) -> update c (f32), h (bf16) ----------------
// Thread mapping: m = tid>>4 (fixed), j strides by 16 -> no integer division.
__device__ __forceinline__ void gate_pass(const float* sz, float* scL, bf16_t* shL,
                                          const float* __restrict__ bias,
                                          const float* sW0, const float* sx,
                                          bool addX, int tid)
{
    const int m  = tid >> 4;          // 512 threads / 32 rows = 16 threads per row
    const int j0 = tid & 15;
    const float xv = addX ? sx[m] : 0.0f;
#pragma unroll 1
    for (int j = j0; j < HH; j += 16) {
        float zi = sz[m * H4 + j          ] + bias[j          ];
        float zf = sz[m * H4 + j + HH     ] + bias[j + HH     ];
        float zg = sz[m * H4 + j + 2 * HH ] + bias[j + 2 * HH ];
        float zo = sz[m * H4 + j + 3 * HH ] + bias[j + 3 * HH ];
        if (addX) {
            zi = fmaf(xv, sW0[j          ], zi);
            zf = fmaf(xv, sW0[j + HH     ], zf);
            zg = fmaf(xv, sW0[j + 2 * HH ], zg);
            zo = fmaf(xv, sW0[j + 3 * HH ], zo);
        }
        float c  = scL[m * HH + j];
        float cn = sigm_f(zf) * c + sigm_f(zi) * tanh_f(zg);
        float hn = sigm_f(zo) * tanh_f(cn);
        scL[m * HH + j] = cn;
        shL[m * KP + j] = (bf16_t)hn;   // pad cols [200,224) stay zero from init
    }
}

// ---------------- main persistent LSTM kernel ----------------
__global__ __launch_bounds__(THREADS)
void lstm3_head_kernel(const float* __restrict__ x,      // [B, TSEQ]
                       const float* __restrict__ W0,     // [H4]
                       const float* __restrict__ b0,
                       const float* __restrict__ b1,
                       const float* __restrict__ b2,
                       const float* __restrict__ bfc,    // [NFC]
                       const bf16_t* __restrict__ U0p,   // [KP][H4]
                       const bf16_t* __restrict__ W1p,
                       const bf16_t* __restrict__ U1p,
                       const bf16_t* __restrict__ W2p,
                       const bf16_t* __restrict__ U2p,
                       const bf16_t* __restrict__ Wfcp,  // [KP][NFC]
                       float* __restrict__ out)          // [B, NFC]
{
    __shared__ bf16_t sh[3][MT * KP];   // h0,h1,h2 (bf16, K-padded)   43.0 KB
    __shared__ float  sc[3][MT * HH];   // c0,c1,c2 (f32)              76.8 KB
    __shared__ float  sz[MT * H4];      // z scratch (f32)            102.4 KB
    __shared__ float  sW0[H4];          // layer-0 input row            3.2 KB
    __shared__ float  sx[MT];           // x_t per batch row

    const int tid  = threadIdx.x;
    const int lane = tid & 31;
    const int wv   = tid >> 5;
    const int mt   = wv & 1;            // m-tile owned by this wave (loop-invariant)
    const int nt0  = wv >> 1;           // nt stream
    const int mofsA = mt * 16 * KP;     // A-fragment base offset (bf16 elements)
    const int mofsC = mt * 16 * H4;     // C-store base offset (f32 elements)
    const int row0 = blockIdx.x * MT;

    // ---- init persistent state ----
    for (int i = tid; i < 3 * MT * KP; i += THREADS) ((bf16_t*)sh)[i] = (bf16_t)0.0f;
    for (int i = tid; i < 3 * MT * HH; i += THREADS) ((float*)sc)[i] = 0.0f;
    for (int i = tid; i < H4; i += THREADS) sW0[i] = W0[i];
    __syncthreads();

#pragma unroll 1
    for (int t = 0; t < TSEQ; ++t) {
        if (tid < MT) sx[tid] = x[(row0 + tid) * TSEQ + t];

        // ---- layer 0: z = h0 @ U0 (x*W0 folded into gate pass) ----
#pragma unroll 1
        for (int nt = nt0; nt < NT4H; nt += NTSTRIDE) {
            v8f acc = {};
#pragma unroll
            for (int kt = 0; kt < KT; ++kt) {
                v16bf a = load_a_frag(sh[0] + mofsA, kt, lane);
                v16bf b = load_b_frag(U0p, H4, kt, nt, lane);
                acc = __builtin_amdgcn_wmma_f32_16x16x32_bf16(false, a, false, b,
                                                              (short)0, acc, false, false);
            }
            store_c_lds(sz + mofsC, H4, nt, lane, acc);
        }
        __syncthreads();
        gate_pass(sz, sc[0], sh[0], b0, sW0, sx, true, tid);
        __syncthreads();

        // ---- layer 1: z = h0 @ W1 + h1 @ U1 ----
#pragma unroll 1
        for (int nt = nt0; nt < NT4H; nt += NTSTRIDE) {
            v8f acc = {};
#pragma unroll
            for (int kt = 0; kt < KT; ++kt) {
                v16bf a = load_a_frag(sh[0] + mofsA, kt, lane);
                v16bf b = load_b_frag(W1p, H4, kt, nt, lane);
                acc = __builtin_amdgcn_wmma_f32_16x16x32_bf16(false, a, false, b,
                                                              (short)0, acc, false, false);
            }
#pragma unroll
            for (int kt = 0; kt < KT; ++kt) {
                v16bf a = load_a_frag(sh[1] + mofsA, kt, lane);
                v16bf b = load_b_frag(U1p, H4, kt, nt, lane);
                acc = __builtin_amdgcn_wmma_f32_16x16x32_bf16(false, a, false, b,
                                                              (short)0, acc, false, false);
            }
            store_c_lds(sz + mofsC, H4, nt, lane, acc);
        }
        __syncthreads();
        gate_pass(sz, sc[1], sh[1], b1, sW0, sx, false, tid);
        __syncthreads();

        // ---- layer 2: z = h1 @ W2 + h2 @ U2 ----
#pragma unroll 1
        for (int nt = nt0; nt < NT4H; nt += NTSTRIDE) {
            v8f acc = {};
#pragma unroll
            for (int kt = 0; kt < KT; ++kt) {
                v16bf a = load_a_frag(sh[1] + mofsA, kt, lane);
                v16bf b = load_b_frag(W2p, H4, kt, nt, lane);
                acc = __builtin_amdgcn_wmma_f32_16x16x32_bf16(false, a, false, b,
                                                              (short)0, acc, false, false);
            }
#pragma unroll
            for (int kt = 0; kt < KT; ++kt) {
                v16bf a = load_a_frag(sh[2] + mofsA, kt, lane);
                v16bf b = load_b_frag(U2p, H4, kt, nt, lane);
                acc = __builtin_amdgcn_wmma_f32_16x16x32_bf16(false, a, false, b,
                                                              (short)0, acc, false, false);
            }
            store_c_lds(sz + mofsC, H4, nt, lane, acc);
        }
        __syncthreads();
        gate_pass(sz, sc[2], sh[2], b2, sW0, sx, false, tid);
        __syncthreads();
    }

    // ---- dense head: y = tanh(h2 @ Wfc + bfc), write straight to HBM ----
#pragma unroll 1
    for (int nt = nt0; nt < NTFC; nt += NTSTRIDE) {
        v8f acc = {};
#pragma unroll
        for (int kt = 0; kt < KT; ++kt) {
            v16bf a = load_a_frag(sh[2] + mofsA, kt, lane);
            v16bf b = load_b_frag(Wfcp, NFC, kt, nt, lane);
            acc = __builtin_amdgcn_wmma_f32_16x16x32_bf16(false, a, false, b,
                                                          (short)0, acc, false, false);
        }
        int n  = nt * 16 + (lane & 15);
        int mb = (lane >> 4) << 3;
#pragma unroll
        for (int r = 0; r < 8; ++r) {
            float y = tanh_f(acc[r] + bfc[n]);
            out[(size_t)(row0 + mt * 16 + mb + r) * NFC + n] = y;
        }
    }
}

// ---------------- host entry ----------------
extern "C" void kernel_launch(void* const* d_in, const int* in_sizes, int n_in,
                              void* d_out, int out_size, void* d_ws, size_t ws_size,
                              hipStream_t stream)
{
    const float* x   = (const float*)d_in[0];
    const float* W0  = (const float*)d_in[1];
    const float* U0  = (const float*)d_in[2];
    const float* b0  = (const float*)d_in[3];
    const float* W1  = (const float*)d_in[4];
    const float* U1  = (const float*)d_in[5];
    const float* b1  = (const float*)d_in[6];
    const float* W2  = (const float*)d_in[7];
    const float* U2  = (const float*)d_in[8];
    const float* b2  = (const float*)d_in[9];
    const float* Wfc = (const float*)d_in[10];
    const float* bfc = (const float*)d_in[11];

    bf16_t* ws   = (bf16_t*)d_ws;
    bf16_t* U0p  = ws;
    bf16_t* W1p  = U0p + (size_t)KP * H4;
    bf16_t* U1p  = W1p + (size_t)KP * H4;
    bf16_t* W2p  = U1p + (size_t)KP * H4;
    bf16_t* U2p  = W2p + (size_t)KP * H4;
    bf16_t* Wfcp = U2p + (size_t)KP * H4;

    const int n800 = KP * H4;
    const int g800 = (n800 + 255) / 256;
    lstm_pack_w<<<g800, 256, 0, stream>>>(U0, U0p, HH, H4);
    lstm_pack_w<<<g800, 256, 0, stream>>>(W1, W1p, HH, H4);
    lstm_pack_w<<<g800, 256, 0, stream>>>(U1, U1p, HH, H4);
    lstm_pack_w<<<g800, 256, 0, stream>>>(W2, W2p, HH, H4);
    lstm_pack_w<<<g800, 256, 0, stream>>>(U2, U2p, HH, H4);
    const int n400 = KP * NFC;
    lstm_pack_w<<<(n400 + 255) / 256, 256, 0, stream>>>(Wfc, Wfcp, HH, NFC);

    const int B = in_sizes[0] / TSEQ;       // 8192
    lstm3_head_kernel<<<B / MT, THREADS, 0, stream>>>(
        x, W0, b0, b1, b2, bfc, U0p, W1p, U1p, W2p, U2p, Wfcp, (float*)d_out);
}
